// LSTM_AE_75359496175947
// MI455X (gfx1250) — compile-verified, matrix operand order checked by reading, not run on hardware
//
#include <hip/hip_runtime.h>
#include <hip/hip_bf16.h>
#include <math.h>

#define NSEQ 16384
#define DF   512
#define G4   2048   // 4 * H

#define ATHENA_USE_TDM 1

typedef __attribute__((ext_vector_type(16))) _Float16 v16h;
typedef __attribute__((ext_vector_type(8)))  _Float16 v8h;
typedef __attribute__((ext_vector_type(8)))  float    v8f;
typedef __attribute__((ext_vector_type(4)))  unsigned u32x4;
typedef __attribute__((ext_vector_type(8)))  int      i32x8;
typedef __attribute__((ext_vector_type(4)))  int      i32x4;

__device__ __forceinline__ float sigm(float x) { return 1.0f / (1.0f + expf(-x)); }

// ---------------------------------------------------------------------------
// Pack: x -> fp16, and build flipped-feature decoder input fxh (last row zero)
// ---------------------------------------------------------------------------
__global__ void pack_x_kernel(const float* __restrict__ x,
                              _Float16* __restrict__ xh,
                              _Float16* __restrict__ fxh) {
  size_t i = (size_t)blockIdx.x * blockDim.x + threadIdx.x;
  if (i >= (size_t)NSEQ * DF) return;
  int t = (int)(i / DF), d = (int)(i % DF);
  float v = x[i];
  xh[i] = (_Float16)v;
  // fxh[t][DF-1-d] = x[t][d]; decoder teacher input row t feeds step t+1.
  fxh[(size_t)t * DF + (DF - 1 - d)] = (t < NSEQ - 1) ? (_Float16)v : (_Float16)0.0f;
}

__global__ void pack_w_kernel(const float* __restrict__ we,
                              const float* __restrict__ wd,
                              _Float16* __restrict__ weh,
                              _Float16* __restrict__ wdh) {
  size_t i = (size_t)blockIdx.x * blockDim.x + threadIdx.x;
  if (i >= (size_t)G4 * DF) return;
  weh[i] = (_Float16)we[i];
  wdh[i] = (_Float16)wd[i];
}

__global__ void pack_b_kernel(const float* __restrict__ bihe, const float* __restrict__ bhhe,
                              const float* __restrict__ bihd, const float* __restrict__ bhhd,
                              float* __restrict__ be, float* __restrict__ bd) {
  int i = blockIdx.x * blockDim.x + threadIdx.x;
  if (i >= G4) return;
  be[i] = bihe[i] + bhhe[i];
  bd[i] = bihd[i] + bhhd[i];
}

// ---------------------------------------------------------------------------
// GEMM: C[m + rowOff][n] = sum_k A[m][k] * W[n][k] + bias[n]
// v_wmma_f32_16x16x32_f16, fragment layouts per CDNA5 ISA 7.12.2.
// ---------------------------------------------------------------------------
__global__ void gemm_xw_kernel(const _Float16* __restrict__ A,
                               const _Float16* __restrict__ W,
                               const float* __restrict__ bias,
                               float* __restrict__ C,
                               int rowOff, int Crows) {
  __shared__ _Float16 Ash[16][DF];   // 16 KB: block's 16 A-rows, full K

  const int tid  = threadIdx.x;
  const int lane = tid & 31;
  const int wv   = tid >> 5;
  const int mBase = blockIdx.x * 16;
  const int nBase = blockIdx.y * 128 + wv * 16;

  for (int i = tid; i < 16 * (DF / 8); i += 256) {
    int r  = i >> 6;
    int c8 = (i & 63) * 8;
    *(v8h*)&Ash[r][c8] = *(const v8h*)&A[(size_t)(mBase + r) * DF + c8];
  }
  __syncthreads();

  const int fr    = lane & 15;        // M for A-frag / N-row for B-frag
  const int hiSel = (lane >> 4) * 8;  // lanes 16-31 take the +8 K slice

  v8f acc = {};
#pragma unroll
  for (int kk = 0; kk < DF / 32; ++kk) {
    const int k0 = kk * 32;
    const _Float16* ap = &Ash[fr][k0 + hiSel];
    v8h alo = *(const v8h*)(ap);
    v8h ahi = *(const v8h*)(ap + 16);
    v16h a  = __builtin_shufflevector(alo, ahi, 0,1,2,3,4,5,6,7,8,9,10,11,12,13,14,15);
    const _Float16* wp = &W[(size_t)(nBase + fr) * DF + k0 + hiSel];
    v8h blo = *(const v8h*)(wp);
    v8h bhi = *(const v8h*)(wp + 16);
    v16h b  = __builtin_shufflevector(blo, bhi, 0,1,2,3,4,5,6,7,8,9,10,11,12,13,14,15);

    acc = __builtin_amdgcn_wmma_f32_16x16x32_f16(
        false, a, false, b, (short)0, acc, false, false);
  }

  const int n  = nBase + fr;
  const float bn = bias ? bias[n] : 0.0f;
  const int mHi  = 8 * (lane >> 4);
#pragma unroll
  for (int i = 0; i < 8; ++i) {
    int row = mBase + i + mHi + rowOff;
    if (row < Crows) C[(size_t)row * G4 + n] = acc[i] + bn;
  }
}

// ---------------------------------------------------------------------------
// Persistent LSTM scan: 16 workgroups (one cluster, one per WGP). Block k owns
// hidden units [k*32, k*32+32). Whh slice (4 gates x 32 units x 512 fp32 =
// 256 KB) resident in the WGP's 320 KB LDS, staged via the Tensor Data Mover.
// Per step: read shared h (double-buffered global), 128x512 matvec, cell
// update for the private 32-unit slice, publish h, cluster barrier.
// ---------------------------------------------------------------------------
__global__ void lstm_scan_kernel(const float* __restrict__ Xg,         // [T][2048] incl. bias
                                 const float* __restrict__ Whh,        // [2048][512] fp32
                                 const _Float16* __restrict__ firstWih,// [2048][512] fp16 or null
                                 const float* __restrict__ firstBias,  // [2048] or null
                                 const float* __restrict__ hInit,      // [512] or null (=0)
                                 const float* __restrict__ cInit,      // [512] or null (=0)
                                 float* __restrict__ hbuf,             // [2][512]
                                 float* __restrict__ outSeq,           // null or [T][512] flipped
                                 float* __restrict__ hnOut,
                                 float* __restrict__ hnOut2,
                                 float* __restrict__ cnOut,
                                 int T) {
  __shared__ float Wsh[128][DF];      // 256 KB fp32 recurrent weights
  __shared__ float hs[DF];
  __shared__ float partial[256];
  __shared__ float gates[128];
  __shared__ float csh[32];

  const int tid = threadIdx.x;
  const int ub  = blockIdx.x * 32;    // first hidden unit owned by this block

#if ATHENA_USE_TDM
  // Stage Whh slice via TDM: one 2D tile per gate block (32 rows x 512 fp32,
  // row stride 512). D# per CDNA5 ISA ch.8.
  if (tid < 32) {
    unsigned ldsBase = (unsigned)(size_t)(&Wsh[0][0]);
    for (int g = 0; g < 4; ++g) {
      unsigned long long ga =
          (unsigned long long)(size_t)(Whh + ((size_t)g * DF + ub) * DF);
      u32x4 g0 = {};
      g0[0] = 1u;                                   // count=1 (valid), user mode
      g0[1] = ldsBase + (unsigned)(g * 32 * DF * 4);// lds_addr (bytes)
      g0[2] = (unsigned)(ga & 0xFFFFFFFFu);         // global_addr[31:0]
      g0[3] = (unsigned)((ga >> 32) & 0x01FFFFFFu)  // global_addr[56:32]
              | (2u << 30);                         // type=2 ("image")
      i32x8 g1 = {};
      g1[0] = (2 << 16);                            // data_size=4B
      g1[1] = (int)((DF & 0xFFFFu) << 16);          // tensor_dim0 lo16 @bits63:48
      g1[2] = (int)((DF >> 16) & 0xFFFF) | (32 << 16); // dim0 hi | tensor_dim1 lo
      g1[3] = 0 | (int)(DF << 16);                  // dim1 hi | tile_dim0=512
      g1[4] = 32;                                   // tile_dim1=32, tile_dim2=0
      g1[5] = DF;                                   // tensor_dim0_stride = 512
      g1[6] = 0;
      g1[7] = 0;
      i32x4 gz = {};
#if defined(__clang_major__) && (__clang_major__ >= 23)
      i32x8 gz8 = {};
      __builtin_amdgcn_tensor_load_to_lds(g0, g1, gz, gz, gz8, 0);
#else
      __builtin_amdgcn_tensor_load_to_lds(g0, g1, gz, gz, 0);
#endif
    }
    __builtin_amdgcn_s_wait_tensorcnt(0);
  }
#else
  for (int i = tid; i < 128 * DF; i += 256) {
    int lr = i >> 9, kk = i & (DF - 1);
    int gr = (lr >> 5) * DF + ub + (lr & 31);
    Wsh[lr][kk] = Whh[(size_t)gr * DF + kk];
  }
#endif

  if (tid < 32) {
    hbuf[ub + tid] = hInit ? hInit[ub + tid] : 0.0f;
    csh[tid]       = cInit ? cInit[ub + tid] : 0.0f;
  }
  __threadfence();
  __syncthreads();
  __builtin_amdgcn_s_cluster_barrier();

  const int lr   = tid & 127;
  const int seg  = tid >> 7;                         // which 256-wide K half
  const int grow = (lr >> 5) * DF + ub + (lr & 31);  // global gate row

  for (int t = 0; t < T; ++t) {
    const int cur = t & 1, nxt = cur ^ 1;

    for (int i = tid; i < DF; i += 256) hs[i] = hbuf[cur * DF + i];
    __syncthreads();

    // Recurrent partial dot: 256 MACs, pure fp32 FMA, ds_load_b128 operands.
    float acc = 0.0f;
    {
      const float4* wr = (const float4*)&Wsh[lr][seg * 256];
      const float4* hr = (const float4*)&hs[seg * 256];
#pragma unroll 4
      for (int i = 0; i < 64; ++i) {
        float4 w = wr[i], h = hr[i];
        acc += w.x * h.x + w.y * h.y + w.z * h.z + w.w * h.w;
      }
    }
    const bool onTheFlyInput = (t == 0) && (firstWih != nullptr);
    if (onTheFlyInput) {
      // decoder step 0: x0 == h0 == h_n, reuse hs for the input-weight dot.
      const _Float16* xr = firstWih + (size_t)grow * DF + seg * 256;
      const float*    hr = &hs[seg * 256];
#pragma unroll 8
      for (int i = 0; i < 256; ++i) acc += (float)xr[i] * hr[i];
      if (seg == 0) acc += firstBias[grow];
    }
    partial[tid] = acc;
    __syncthreads();

    if (tid < 128) {
      float g = partial[tid] + partial[tid + 128];
      if (!onTheFlyInput) g += Xg[(size_t)t * G4 + grow];
      gates[tid] = g;
      if (t + 1 < T)
        __builtin_prefetch(&Xg[(size_t)(t + 1) * G4 + grow], 0, 1);
    }
    __syncthreads();

    if (tid < 32) {
      float ig = sigm(gates[tid]);
      float fg = sigm(gates[32 + tid]);
      float gg = tanhf(gates[64 + tid]);
      float og = sigm(gates[96 + tid]);
      float c  = fg * csh[tid] + ig * gg;
      csh[tid] = c;
      float h  = og * tanhf(c);
      hbuf[nxt * DF + ub + tid] = h;
      if (outSeq) outSeq[(size_t)t * DF + (DF - 1 - (ub + tid))] = h;  // feature flip
    }
    __threadfence();
    __syncthreads();
    __builtin_amdgcn_s_cluster_barrier();
  }

  if (tid < 32) {
    float hf = hbuf[(T & 1) * DF + ub + tid];
    if (hnOut)  hnOut[ub + tid]  = hf;
    if (hnOut2) hnOut2[ub + tid] = hf;
    if (cnOut)  cnOut[ub + tid]  = csh[tid];
  }
}

// ---------------------------------------------------------------------------
// Launch
// ---------------------------------------------------------------------------
extern "C" void kernel_launch(void* const* d_in, const int* in_sizes, int n_in,
                              void* d_out, int out_size, void* d_ws, size_t ws_size,
                              hipStream_t stream) {
  (void)in_sizes; (void)n_in; (void)out_size; (void)ws_size;

  const float* x     = (const float*)d_in[0];
  const float* Wih_e = (const float*)d_in[1];
  const float* Whh_e = (const float*)d_in[2];
  const float* bih_e = (const float*)d_in[3];
  const float* bhh_e = (const float*)d_in[4];
  const float* Wih_d = (const float*)d_in[5];
  const float* Whh_d = (const float*)d_in[6];
  const float* bih_d = (const float*)d_in[7];
  const float* bhh_d = (const float*)d_in[8];
  float* out = (float*)d_out;

  char* ws = (char*)d_ws;
  size_t o = 0;
  _Float16* xh  = (_Float16*)(ws + o); o += (size_t)NSEQ * DF * sizeof(_Float16);
  _Float16* fxh = (_Float16*)(ws + o); o += (size_t)NSEQ * DF * sizeof(_Float16);
  _Float16* wie = (_Float16*)(ws + o); o += (size_t)G4 * DF * sizeof(_Float16);
  _Float16* wid = (_Float16*)(ws + o); o += (size_t)G4 * DF * sizeof(_Float16);
  float* be   = (float*)(ws + o); o += (size_t)G4 * sizeof(float);
  float* bd   = (float*)(ws + o); o += (size_t)G4 * sizeof(float);
  float* Xge  = (float*)(ws + o); o += (size_t)NSEQ * G4 * sizeof(float);
  float* Xgd  = (float*)(ws + o); o += (size_t)NSEQ * G4 * sizeof(float);
  float* hbuf = (float*)(ws + o); o += (size_t)2 * DF * sizeof(float);
  float* hn   = (float*)(ws + o); o += (size_t)DF * sizeof(float);
  float* cn   = (float*)(ws + o); o += (size_t)DF * sizeof(float);

  {
    size_t n = (size_t)NSEQ * DF;
    pack_x_kernel<<<(unsigned)((n + 255) / 256), 256, 0, stream>>>(x, xh, fxh);
  }
  {
    size_t n = (size_t)G4 * DF;
    pack_w_kernel<<<(unsigned)((n + 255) / 256), 256, 0, stream>>>(Wih_e, Wih_d, wie, wid);
  }
  pack_b_kernel<<<(G4 + 255) / 256, 256, 0, stream>>>(bih_e, bhh_e, bih_d, bhh_d, be, bd);

  dim3 ggrid(NSEQ / 16, G4 / 128);
  gemm_xw_kernel<<<ggrid, 256, 0, stream>>>(xh,  wie, be, Xge, /*rowOff=*/0, NSEQ);
  gemm_xw_kernel<<<ggrid, 256, 0, stream>>>(fxh, wid, bd, Xgd, /*rowOff=*/1, NSEQ);

  lstm_scan_kernel<<<16, 256, 0, stream>>>(Xge, Whh_e,
                                           nullptr, nullptr,
                                           nullptr, nullptr,
                                           hbuf, /*outSeq=*/nullptr,
                                           hn, out, cn, NSEQ);

  lstm_scan_kernel<<<16, 256, 0, stream>>>(Xgd, Whh_d,
                                           wid, bd,
                                           hn, cn,
                                           hbuf, out + DF,
                                           nullptr, nullptr, nullptr, NSEQ);
}
